// MultiHeadProjection_3564822855692
// MI455X (gfx1250) — matrix operand order repaired and checked by reading, not compiled, not run on hardware
//
#include <hip/hip_runtime.h>

// MI455X / gfx1250, wave32. bf16x3 split-precision GEMM via v_wmma_f32_16x16x32_bf16.
// q/k/v projections: C[8192,1024] = A[8192,1024] * W[1024,1024]^T, scattered to [B,H,S,D].
// 256x256 workgroup tile, 512 threads = 16 waves (4x4), 64x64 per wave.
// Weights pre-split (hi/lo bf16) into d_ws once; streamed into LDS with
// global_load_async_to_lds_b128 (ASYNCcnt). Activations split in-loop via v_perm_b32.

typedef __attribute__((ext_vector_type(16))) __bf16 v16bf;
typedef __attribute__((ext_vector_type(8)))  float  v8f;

#define E_DIM   1024
#define M_TOT   8192            // B*S = 4*2048
#define N_TOT   1024            // H*D = 16*64
#define OUT_PER 8388608         // B*H*S*D per tensor
#define W_ELEMS (N_TOT * E_DIM) // 1M elements per weight tensor

#define TM 256
#define TN 256
#define TK 32
#define NTHREADS 512
#define PITCH 40                // ushorts per LDS row: 32 data + 8 pad (80B, 16B-aligned, conflict-free)

// Split one f32 pair into packed hi-bf16 pair and lo-bf16 (residual) pair.
// v_perm_b32 packs the two high halves in one instruction.
__device__ __forceinline__ void split2(float a, float b, unsigned int& h, unsigned int& l) {
    union F { float f; unsigned int u; };
    F ca, cb; ca.f = a; cb.f = b;
    h = __builtin_amdgcn_perm(cb.u, ca.u, 0x07060302u);   // {hi16(b), hi16(a)}
    F rha, rhb; rha.u = ca.u & 0xffff0000u; rhb.u = cb.u & 0xffff0000u;
    F cla, clb; cla.f = a - rha.f; clb.f = b - rhb.f;     // residuals
    l = __builtin_amdgcn_perm(clb.u, cla.u, 0x07060302u);
}

// Load a 16-lane-striped 16-bf16 fragment (two 16B chunks) from LDS.
__device__ __forceinline__ v16bf ldfrag(const unsigned short* base, int row, int off0, int off1) {
    union { uint4 u[2]; v16bf v; } t;
    const unsigned short* p = base + row * PITCH;
    t.u[0] = *(const uint4*)(p + off0);
    t.u[1] = *(const uint4*)(p + off1);
    return t.v;
}

// ---------------- pass 1: split weights once into workspace --------------------
__global__ __launch_bounds__(256) void wsplit_kernel(
    const float* __restrict__ w0, const float* __restrict__ w1, const float* __restrict__ w2,
    unsigned short* __restrict__ whi, unsigned short* __restrict__ wlo)
{
    const int z = blockIdx.y;
    const float* W = (z == 0) ? w0 : (z == 1) ? w1 : w2;
    size_t base = ((size_t)blockIdx.x * 256 + threadIdx.x) * 4;
    float4 f = *(const float4*)(W + base);
    unsigned int h01, l01, h23, l23;
    split2(f.x, f.y, h01, l01);
    split2(f.z, f.w, h23, l23);
    size_t o = (size_t)z * W_ELEMS + base;
    *(uint2*)(whi + o) = make_uint2(h01, h23);
    *(uint2*)(wlo + o) = make_uint2(l01, l23);
}

// ---------------- pass 2: GEMM ------------------------------------------------
template <bool PRESPLIT>
__global__ __launch_bounds__(NTHREADS) void mhp_bf16x3_kernel(
    const float* __restrict__ q,  const float* __restrict__ k,  const float* __restrict__ v,
    const float* __restrict__ wq, const float* __restrict__ wk, const float* __restrict__ wv,
    const unsigned short* __restrict__ whi, const unsigned short* __restrict__ wlo,
    float* __restrict__ out)
{
    __shared__ unsigned short sA[2][2][TM * PITCH];   // [buf][hi/lo]  80 KB
    __shared__ unsigned short sW[2][2][TN * PITCH];   //               80 KB

    const int z = blockIdx.z;
    const float* A = (z == 0) ? q  : (z == 1) ? k  : v;
    const float* W = (z == 0) ? wq : (z == 1) ? wk : wv;
    const unsigned short* Whi = whi + (size_t)z * W_ELEMS;
    const unsigned short* Wlo = wlo + (size_t)z * W_ELEMS;
    float* O = out + (size_t)z * (size_t)OUT_PER;

    const int tid    = threadIdx.x;
    const int wid    = tid >> 5;
    const int lane   = tid & 31;
    const int laneLo = lane & 15;
    const int laneHi = lane >> 4;
    const int waveM  = wid >> 2;          // 0..3 -> 64 rows each
    const int waveN  = wid & 3;           // 0..3 -> 64 cols each

    const int blockM = blockIdx.x * TM;
    const int blockN = blockIdx.y * TN;

    // ---- staging helpers ---------------------------------------------------
    auto gloadA = [&](int kBase, float4* ra) {
        #pragma unroll
        for (int j = 0; j < 4; ++j) {
            int i = tid + NTHREADS * j;         // 0..2047
            int r = i >> 3;                     // 0..255
            int c = (i & 7) << 2;               // 0,4,...,28
            ra[j] = *(const float4*)(A + (size_t)(blockM + r) * E_DIM + kBase + c);
        }
    };
    auto stageA = [&](int buf, const float4* ra) {
        #pragma unroll
        for (int j = 0; j < 4; ++j) {
            int i = tid + NTHREADS * j;
            int r = i >> 3;
            int c = (i & 7) << 2;
            unsigned int h01, l01, h23, l23;
            split2(ra[j].x, ra[j].y, h01, l01);
            split2(ra[j].z, ra[j].w, h23, l23);
            *(uint2*)&sA[buf][0][r * PITCH + c] = make_uint2(h01, h23);
            *(uint2*)&sA[buf][1][r * PITCH + c] = make_uint2(l01, l23);
        }
    };
    // Fallback (no workspace): load f32 W and split in-loop.
    auto gloadW = [&](int kBase, float4* rw) {
        #pragma unroll
        for (int j = 0; j < 4; ++j) {
            int i = tid + NTHREADS * j;
            int r = i >> 3;
            int c = (i & 7) << 2;
            rw[j] = *(const float4*)(W + (size_t)(blockN + r) * E_DIM + kBase + c);
        }
    };
    auto stageW = [&](int buf, const float4* rw) {
        #pragma unroll
        for (int j = 0; j < 4; ++j) {
            int i = tid + NTHREADS * j;
            int r = i >> 3;
            int c = (i & 7) << 2;
            unsigned int h01, l01, h23, l23;
            split2(rw[j].x, rw[j].y, h01, l01);
            split2(rw[j].z, rw[j].w, h23, l23);
            *(uint2*)&sW[buf][0][r * PITCH + c] = make_uint2(h01, h23);
            *(uint2*)&sW[buf][1][r * PITCH + c] = make_uint2(l01, l23);
        }
    };
    // Pre-split path: async DMA bf16 weight tiles straight into LDS (ASYNCcnt).
    auto wstageAsync = [&](int buf, int kBase) {
        #pragma unroll
        for (int j = 0; j < 2; ++j) {
            int i = tid + NTHREADS * j;         // 0..1023
            int r = i >> 2;                     // 0..255
            int c = (i & 3) << 3;               // 0,8,16,24 (ushorts)
            const unsigned short* gH = Whi + (size_t)(blockN + r) * E_DIM + kBase + c;
            const unsigned short* gL = Wlo + (size_t)(blockN + r) * E_DIM + kBase + c;
            unsigned int dH = (unsigned int)(size_t)&sW[buf][0][r * PITCH + c];
            unsigned int dL = (unsigned int)(size_t)&sW[buf][1][r * PITCH + c];
            asm volatile("global_load_async_to_lds_b128 %0, %1, off"
                         :: "v"(dH), "v"(gH) : "memory");
            asm volatile("global_load_async_to_lds_b128 %0, %1, off"
                         :: "v"(dL), "v"(gL) : "memory");
        }
    };

    v8f acc[4][4] = {};

    // ---- software-pipelined K loop ----------------------------------------
    float4 ra[4], rw[4];
    if (PRESPLIT) {
        wstageAsync(0, 0);
    } else {
        gloadW(0, rw);
    }
    gloadA(0, ra);
    stageA(0, ra);
    if (!PRESPLIT) stageW(0, rw);
    if (PRESPLIT) asm volatile("s_wait_asynccnt 0" ::: "memory");
    __syncthreads();

    int cur = 0;
    const int NSTAGE = E_DIM / TK;            // 32
    for (int kt = 0; kt < NSTAGE; ++kt) {
        const bool more = (kt + 1 < NSTAGE);
        if (PRESPLIT && more) wstageAsync(cur ^ 1, (kt + 1) * TK);

        float4 na[4], nw[4];
        if (more) {
            gloadA((kt + 1) * TK, na);
            if (!PRESPLIT) gloadW((kt + 1) * TK, nw);
        }

        const unsigned short* aHiB = sA[cur][0];
        const unsigned short* aLoB = sA[cur][1];
        const unsigned short* wHiB = sW[cur][0];
        const unsigned short* wLoB = sW[cur][1];

        // B fragments resident for the whole stage: K in {0..7,16..23} / {8..15,24..31} by lane-half
        v16bf bHi[4], bLo[4];
        #pragma unroll
        for (int ns = 0; ns < 4; ++ns) {
            int nrow = waveN * 64 + ns * 16 + laneLo;
            bHi[ns] = ldfrag(wHiB, nrow, laneHi * 8, laneHi * 8 + 16);
            bLo[ns] = ldfrag(wLoB, nrow, laneHi * 8, laneHi * 8 + 16);
        }
        #pragma unroll
        for (int ms = 0; ms < 4; ++ms) {
            int mrow = waveM * 64 + ms * 16 + laneLo;
            // A fragments: K in {0..7,16..23} / {8..15,24..31} by lane-half
            v16bf aHi = ldfrag(aHiB, mrow, laneHi * 8, laneHi * 8 + 16);
            v16bf aLo = ldfrag(aLoB, mrow, laneHi * 8, laneHi * 8 + 16);
            #pragma unroll
            for (int ns = 0; ns < 4; ++ns) {
                acc[ms][ns] = __builtin_amdgcn_wmma_f32_16x16x32_bf16(
                    false, aHi, false, bHi[ns], (short)0, acc[ms][ns], false, false);
                acc[ms][ns] = __builtin_amdgcn_wmma_f32_16x16x32_bf16(
                    false, aHi, false, bLo[ns], (short)0, acc[ms][ns], false, false);
                acc[ms][ns] = __builtin_amdgcn_wmma_f32_16x16x32_bf16(
                    false, aLo, false, bHi[ns], (short)0, acc[ms][ns], false, false);
            }
        }

        if (more) {
            stageA(cur ^ 1, na);
            if (!PRESPLIT) stageW(cur ^ 1, nw);
        }
        if (PRESPLIT) asm volatile("s_wait_asynccnt 0" ::: "memory");
        __syncthreads();
        cur ^= 1;
    }

    // ---- epilogue: scatter C[16x16] tiles into [B,H,S,D] -------------------
    // C layout: VGPR r holds (M = r + 8*laneHi, N = laneLo)
    #pragma unroll
    for (int ms = 0; ms < 4; ++ms) {
        #pragma unroll
        for (int ns = 0; ns < 4; ++ns) {
            int nBase = blockN + waveN * 64 + ns * 16 + laneLo;   // = h*64 + d
            int h = nBase >> 6, d = nBase & 63;
            #pragma unroll
            for (int r = 0; r < 8; ++r) {
                int m = blockM + waveM * 64 + ms * 16 + laneHi * 8 + r; // = b*2048 + s
                int b = m >> 11, s = m & 2047;
                size_t o = ((((size_t)b * 16 + h) * 2048 + s) * 64 + d);
                O[o] = acc[ms][ns][r];
            }
        }
    }
}

extern "C" void kernel_launch(void* const* d_in, const int* in_sizes, int n_in,
                              void* d_out, int out_size, void* d_ws, size_t ws_size,
                              hipStream_t stream) {
    const float* q  = (const float*)d_in[0];
    const float* k  = (const float*)d_in[1];
    const float* v  = (const float*)d_in[2];
    const float* wq = (const float*)d_in[3];
    const float* wk = (const float*)d_in[4];
    const float* wv = (const float*)d_in[5];
    float* out = (float*)d_out;

    const size_t WS_NEED = (size_t)3 * W_ELEMS * 2 * sizeof(unsigned short); // 12 MiB
    dim3 grid(M_TOT / TM, N_TOT / TN, 3);   // 32 x 4 x 3

    if (ws_size >= WS_NEED) {
        unsigned short* whi = (unsigned short*)d_ws;
        unsigned short* wlo = whi + (size_t)3 * W_ELEMS;
        dim3 sgrid(W_ELEMS / 4 / 256, 3);   // 1024 x 3
        wsplit_kernel<<<sgrid, 256, 0, stream>>>(wq, wk, wv, whi, wlo);
        mhp_bf16x3_kernel<true><<<grid, NTHREADS, 0, stream>>>(
            q, k, v, wq, wk, wv, whi, wlo, out);
    } else {
        mhp_bf16x3_kernel<false><<<grid, NTHREADS, 0, stream>>>(
            q, k, v, wq, wk, wv, (const unsigned short*)nullptr,
            (const unsigned short*)nullptr, out);
    }
}
